// DSNet_8564164788861
// MI455X (gfx1250) — matrix operand order repaired and checked by reading, not compiled, not run on hardware
//
#include <hip/hip_runtime.h>
#include <hip/hip_bf16.h>

typedef __bf16 bf16_t;
typedef __attribute__((ext_vector_type(16))) __bf16 v16bf;
typedef __attribute__((ext_vector_type(8)))  __bf16 v8bf;
typedef __attribute__((ext_vector_type(4)))  __bf16 v4bf;
typedef __attribute__((ext_vector_type(8)))  float  v8f;

#define D_IN   64
#define D_SEQ  512
#define D_H    128
#define D_OUT  64
#define T_REF  32768
#define N_SPLITS     32768
#define N_REF_SPLITS 16384

// Workspace layout (bytes)
#define WS_S     0                                       // 32768*512 bf16 = 32 MB
#define WS_WEFF  ((size_t)N_SPLITS * D_SEQ * 2)          // 512*256 bf16
#define WS_BETA  (WS_WEFF + (size_t)D_SEQ * 256 * 2)     // 512 f32
#define WS_W2B   (WS_BETA + (size_t)D_SEQ * 4)           // 128*512 bf16
#define WS_W3B   (WS_W2B + (size_t)D_H * D_SEQ * 2)      // 64*128 bf16

#define BT_STRIDE 264   // 128 n-rows x 256 K, padded: 528B row -> conflict-free
#define H_STRIDE  136   // 16 rows x 128 cols, padded: 272B row -> conflict-free

__device__ __forceinline__ v8f vzero8() {
  v8f z;
  #pragma unroll
  for (int i = 0; i < 8; ++i) z[i] = 0.0f;
  return z;
}

// A fragment: row-major [M][K] bf16 matrix, base at (tileRow0, kPanel0).
// lane<16: row=lane, K {0..7,16..23}; lane>=16: row=lane-16, K {8..15,24..31}.
__device__ __forceinline__ v16bf load_frag_a(const bf16_t* base, int row_stride) {
  int lane = threadIdx.x & 31;
  int r = lane & 15;
  int kof = (lane & 16) ? 8 : 0;
  const bf16_t* p = base + (size_t)r * row_stride + kof;
  v8bf lo = *(const v8bf*)(p);
  v8bf hi = *(const v8bf*)(p + 16);
  return __builtin_shufflevector(lo, hi, 0,1,2,3,4,5,6,7,8,9,10,11,12,13,14,15);
}

// B fragment: B[k][n] = W[n][k], W row-major [N][K] bf16, base at (nTile0, kPanel0).
// lane<16: n=lane, K {0..7,16..23}; lane>=16: n=lane-16, K {8..15,24..31}.
__device__ __forceinline__ v16bf load_frag_b(const bf16_t* base, int row_stride) {
  int lane = threadIdx.x & 31;
  int n = lane & 15;
  int kof = (lane & 16) ? 8 : 0;
  const bf16_t* p = base + (size_t)n * row_stride + kof;
  v8bf lo = *(const v8bf*)(p);
  v8bf hi = *(const v8bf*)(p + 16);
  return __builtin_shufflevector(lo, hi, 0,1,2,3,4,5,6,7,8,9,10,11,12,13,14,15);
}

__device__ __forceinline__ v8f wmma_bf16(v16bf a, v16bf b, v8f c) {
  return __builtin_amdgcn_wmma_f32_16x16x32_bf16(false, a, false, b, (short)0, c,
                                                 false, false);
}

// ---------------------------------------------------------------------------
// Prep: Weff[l, i*4+d] = sum_{kk, c, k1=d-c} w1[c,kk]*w_seq[l+kk-1, i, k1]
//       beta[l] = b1 + sum_kk (w1[0,kk]+w1[1,kk]) * b_seq[l+kk-1]
//       W2b = bf16(w2[:,:,1]), W3b = bf16(w3[:,:,1])
// ---------------------------------------------------------------------------
__global__ void k_prep(const float* __restrict__ w_seq, const float* __restrict__ b_seq,
                       const float* __restrict__ w1, const float* __restrict__ b1,
                       const float* __restrict__ w2, const float* __restrict__ w3,
                       bf16_t* __restrict__ Weff, float* __restrict__ beta,
                       bf16_t* __restrict__ W2b, bf16_t* __restrict__ W3b) {
  int b = blockIdx.x;
  int t = threadIdx.x;
  if (b < 512) {
    int l = b;
    int i = t >> 2;
    int d = t & 3;
    float sum = 0.0f;
    #pragma unroll
    for (int kk = 0; kk < 3; ++kk) {
      int lch = l + kk - 1;
      if (lch >= 0 && lch < D_SEQ) {
        #pragma unroll
        for (int c = 0; c < 2; ++c) {
          int k1 = d - c;
          if (k1 >= 0 && k1 < 3)
            sum += w1[c * 3 + kk] * w_seq[((size_t)lch * D_IN + i) * 3 + k1];
        }
      }
    }
    Weff[(size_t)l * 256 + i * 4 + d] = (bf16_t)sum;
    if (t == 0) {
      float bsum = b1[0];
      #pragma unroll
      for (int kk = 0; kk < 3; ++kk) {
        int lch = l + kk - 1;
        if (lch >= 0 && lch < D_SEQ)
          bsum += (w1[kk] + w1[3 + kk]) * b_seq[lch];
      }
      beta[l] = bsum;
    }
  } else if (b < 768) {
    int idx = (b - 512) * 256 + t;                 // 0..65535
    int n = idx >> 9, k = idx & 511;
    W2b[idx] = (bf16_t)w2[((size_t)n * D_SEQ + k) * 3 + 1];
  } else {
    int idx = (b - 768) * 256 + t;                 // 0..8191
    int n = idx >> 7, k = idx & 127;
    W3b[idx] = (bf16_t)w3[((size_t)n * D_H + k) * 3 + 1];
  }
}

// ---------------------------------------------------------------------------
// S rows [16384, 32768): 6-tap linear1 on adjacent x rows (memory-bound, VALU)
// ---------------------------------------------------------------------------
__global__ void k_s1x(const float* __restrict__ x, const float* __restrict__ w1,
                      const float* __restrict__ b1, bf16_t* __restrict__ S) {
  int blk = blockIdx.x;                 // 0..16383
  int n = N_REF_SPLITS + blk;
  const float* x0 = x + (size_t)(2 * blk) * D_SEQ;
  const float* x1 = x0 + D_SEQ;
  float w[6];
  #pragma unroll
  for (int i = 0; i < 6; ++i) w[i] = w1[i];
  float bias = b1[0];
  for (int l = threadIdx.x; l < D_SEQ; l += blockDim.x) {
    float s = bias;
    #pragma unroll
    for (int kk = 0; kk < 3; ++kk) {
      int lch = l + kk - 1;
      if (lch >= 0 && lch < D_SEQ)
        s += w[kk] * x0[lch] + w[3 + kk] * x1[lch];
    }
    S[(size_t)n * D_SEQ + l] = (bf16_t)fmaxf(s, 0.0f);
  }
}

// ---------------------------------------------------------------------------
// S rows [0, 16384): WMMA GEMM  S[n,l] = relu(beta[l] + Weff(512x256) * im2col(refer))
// Block covers 128 n. Stage the im2col slice of refer into LDS (bf16, padded
// rows) once; boundary masking only in first/last block. Each of 8 waves owns
// a 16-wide n tile and sweeps all 512 l in 8 passes of 4 l-tiles (acc = 32
// VGPRs, A-frag batch = 32 VGPRs -> no spills).
// ---------------------------------------------------------------------------
__global__ void __launch_bounds__(256, 1)
k_s1ref(const float* __restrict__ refer,
        const bf16_t* __restrict__ Weff,
        const float* __restrict__ beta,
        bf16_t* __restrict__ S) {
  __shared__ bf16_t Bt[128 * BT_STRIDE];

  int tid = threadIdx.x;
  int nblk0 = blockIdx.x * 128;
  bool interior = (blockIdx.x != 0) && (blockIdx.x != (int)gridDim.x - 1);

  // Stage-in: Bt[nloc][i*4+d] = refer[i, 2*(nblk0+nloc)-1+d] (0 outside [0,T_REF))
  for (int idx = tid; idx < 128 * D_IN; idx += 256) {
    int nloc = idx >> 6;
    int i = idx & 63;
    long tbase = 2L * (nblk0 + nloc) - 1;
    const float* rp = refer + (size_t)i * T_REF + tbase;
    float f0, f1, f2, f3;
    if (interior) {
      f0 = rp[0]; f1 = rp[1]; f2 = rp[2]; f3 = rp[3];
    } else {
      f0 = (tbase >= 0)        ? rp[0] : 0.0f;   // only n==0, d==0
      f1 = rp[1];                                 // 2n in [0, T_REF) always
      f2 = rp[2];                                 // 2n+1 in [0, T_REF) always
      f3 = (tbase + 3 < T_REF) ? rp[3] : 0.0f;   // only n==16383, d==3
    }
    v4bf pk;
    pk[0] = (bf16_t)f0; pk[1] = (bf16_t)f1; pk[2] = (bf16_t)f2; pk[3] = (bf16_t)f3;
    *(v4bf*)&Bt[(size_t)nloc * BT_STRIDE + i * 4] = pk;
  }
  __syncthreads();

  int wave = tid >> 5;
  int lane = tid & 31;
  int hi = (lane & 16) ? 1 : 0;
  int colN = lane & 15;
  int nloc0 = wave * 16;
  int n0 = nblk0 + nloc0;

  #pragma unroll 1
  for (int pass = 0; pass < 8; ++pass) {
    int l0 = pass * 64;
    v8f acc[4];
    #pragma unroll
    for (int a = 0; a < 4; ++a) acc[a] = vzero8();

    #pragma unroll 1
    for (int kp = 0; kp < 8; ++kp) {    // K = 256 in panels of 32
      v16bf bfrag = load_frag_b(&Bt[(size_t)nloc0 * BT_STRIDE + kp * 32], BT_STRIDE);
      v16bf af[4];
      #pragma unroll
      for (int lt = 0; lt < 4; ++lt)
        af[lt] = load_frag_a(Weff + (size_t)(l0 + lt * 16) * 256 + kp * 32, 256);
      #pragma unroll
      for (int lt = 0; lt < 4; ++lt)
        acc[lt] = wmma_bf16(af[lt], bfrag, acc[lt]);
    }
    // epilogue: element j -> (l = l0+lt*16+8*hi+j, n = n0+colN); 8 consecutive l
    #pragma unroll
    for (int lt = 0; lt < 4; ++lt) {
      int lbase = l0 + lt * 16 + hi * 8;
      v8bf pk;
      #pragma unroll
      for (int j = 0; j < 8; ++j)
        pk[j] = (bf16_t)fmaxf(acc[lt][j] + beta[lbase + j], 0.0f);
      *(v8bf*)&S[(size_t)(n0 + colN) * D_SEQ + lbase] = pk;
    }
  }
}

// ---------------------------------------------------------------------------
// Fused GEMM1+GEMM2: H = relu(S*W2^T + b2) (K=512, N=128) kept in LDS as bf16,
// then O = sigmoid(H*W3^T + b3) (K=128, N=64). Block = 128 rows, wave = 16 rows.
// B-frags loaded in groups of 4 to bound register pressure (acc 64 + bf 32).
// ---------------------------------------------------------------------------
__global__ void __launch_bounds__(256, 1)
k_gemm12(const bf16_t* __restrict__ S,
         const bf16_t* __restrict__ W2b, const float* __restrict__ b2,
         const bf16_t* __restrict__ W3b, const float* __restrict__ b3,
         float* __restrict__ out) {
  __shared__ bf16_t ldsH[8][16 * H_STRIDE];
  int wave = threadIdx.x >> 5;
  int lane = threadIdx.x & 31;
  int hi = (lane & 16) ? 1 : 0;
  int colN = lane & 15;
  int m0 = blockIdx.x * 128 + wave * 16;

  // GEMM1: 16 x 128, K = 512
  v8f acc[8];
  #pragma unroll
  for (int a = 0; a < 8; ++a) acc[a] = vzero8();
  #pragma unroll 2
  for (int kp = 0; kp < 16; ++kp) {
    v16bf afrag = load_frag_a(S + (size_t)m0 * D_SEQ + kp * 32, D_SEQ);
    #pragma unroll
    for (int g = 0; g < 2; ++g) {
      v16bf bf[4];
      #pragma unroll
      for (int nt = 0; nt < 4; ++nt)
        bf[nt] = load_frag_b(W2b + (size_t)((g * 4 + nt) * 16) * D_SEQ + kp * 32, D_SEQ);
      #pragma unroll
      for (int nt = 0; nt < 4; ++nt)
        acc[g * 4 + nt] = wmma_bf16(afrag, bf[nt], acc[g * 4 + nt]);
    }
  }
  // epilogue 1: relu + bias -> LDS strip (16 x 128 bf16, padded rows)
  #pragma unroll
  for (int nt = 0; nt < 8; ++nt) {
    int col = nt * 16 + colN;
    float bias = b2[col];
    #pragma unroll
    for (int j = 0; j < 8; ++j) {
      int row = j + hi * 8;
      ldsH[wave][row * H_STRIDE + col] = (bf16_t)fmaxf(acc[nt][j] + bias, 0.0f);
    }
  }

  // GEMM2: 16 x 64, K = 128, A from LDS (same wave wrote it; DScnt orders it)
  v8f acc2[4];
  #pragma unroll
  for (int a = 0; a < 4; ++a) acc2[a] = vzero8();
  #pragma unroll
  for (int kp = 0; kp < 4; ++kp) {
    v16bf afrag = load_frag_a(&ldsH[wave][kp * 32], H_STRIDE);
    v16bf bf2[4];
    #pragma unroll
    for (int nt = 0; nt < 4; ++nt)
      bf2[nt] = load_frag_b(W3b + (size_t)(nt * 16) * D_H + kp * 32, D_H);
    #pragma unroll
    for (int nt = 0; nt < 4; ++nt)
      acc2[nt] = wmma_bf16(afrag, bf2[nt], acc2[nt]);
  }
  // epilogue 2: sigmoid + store fp32 output (n_splits, 64, 1) flat
  #pragma unroll
  for (int nt = 0; nt < 4; ++nt) {
    int col = nt * 16 + colN;
    float bias = b3[col];
    #pragma unroll
    for (int j = 0; j < 8; ++j) {
      int m = m0 + j + hi * 8;
      float v = acc2[nt][j] + bias;
      out[(size_t)m * D_OUT + col] = 1.0f / (1.0f + __expf(-v));
    }
  }
}

extern "C" void kernel_launch(void* const* d_in, const int* in_sizes, int n_in,
                              void* d_out, int out_size, void* d_ws, size_t ws_size,
                              hipStream_t stream) {
  const float* refer = (const float*)d_in[0];
  const float* x     = (const float*)d_in[1];
  const float* w_seq = (const float*)d_in[2];
  const float* b_seq = (const float*)d_in[3];
  const float* w1    = (const float*)d_in[4];
  const float* b1    = (const float*)d_in[5];
  const float* w2    = (const float*)d_in[6];
  const float* b2    = (const float*)d_in[7];
  const float* w3    = (const float*)d_in[8];
  const float* b3    = (const float*)d_in[9];

  char* ws = (char*)d_ws;
  bf16_t* S    = (bf16_t*)(ws + WS_S);
  bf16_t* Weff = (bf16_t*)(ws + WS_WEFF);
  float*  beta = (float*)(ws + WS_BETA);
  bf16_t* W2b  = (bf16_t*)(ws + WS_W2B);
  bf16_t* W3b  = (bf16_t*)(ws + WS_W3B);

  k_prep<<<800, 256, 0, stream>>>(w_seq, b_seq, w1, b1, w2, w3, Weff, beta, W2b, W3b);
  k_s1x<<<N_REF_SPLITS, 256, 0, stream>>>(x, w1, b1, S);
  k_s1ref<<<N_REF_SPLITS / 128, 256, 0, stream>>>(refer, Weff, beta, S);
  k_gemm12<<<N_SPLITS / 128, 256, 0, stream>>>(S, W2b, b2, W3b, b3, (float*)d_out);
}
